// Attention_75514114998541
// MI455X (gfx1250) — matrix operand order repaired and checked by reading, not compiled, not run on hardware
//
#include <hip/hip_runtime.h>
#include <hip/hip_bf16.h>

// Problem constants (match reference)
#define HIDDEN 2048
#define NHEADS 16
#define NKV 4
#define HDIM 128
#define SEQ 2048
#define NB 2

typedef __attribute__((ext_vector_type(16))) __bf16        v16bf;
typedef __attribute__((ext_vector_type(8)))  float         v8f;
typedef __attribute__((ext_vector_type(8)))  unsigned int  v8u;
typedef int v4i __attribute__((vector_size(16)));  // builtin's pointee type

// ---- gfx1250 async global->LDS path (ASYNCcnt), with safe fallback --------
#if defined(__has_builtin)
#if __has_builtin(__builtin_amdgcn_global_load_async_to_lds_b128)
#define HAVE_ASYNC_LDS 1
#endif
#if __has_builtin(__builtin_amdgcn_s_wait_asynccnt)
#define HAVE_WAIT_ASYNC 1
#endif
#endif
#ifndef HAVE_ASYNC_LDS
#define HAVE_ASYNC_LDS 0
#endif
#ifndef HAVE_WAIT_ASYNC
#define HAVE_WAIT_ASYNC 0
#endif

static __device__ __forceinline__ void copy16(const unsigned short* g, unsigned short* l) {
#if HAVE_ASYNC_LDS
  // (global src, LDS dst, imm offset, imm cpol); per-lane addresses.
  __builtin_amdgcn_global_load_async_to_lds_b128(
      (__attribute__((address_space(1))) v4i*)g,
      (__attribute__((address_space(3))) v4i*)l, 0, 0);
#else
  *(uint4*)l = *(const uint4*)g;
#endif
}
static __device__ __forceinline__ void async_join() {
#if HAVE_ASYNC_LDS
#if HAVE_WAIT_ASYNC
  __builtin_amdgcn_s_wait_asynccnt(0);
#else
  asm volatile("s_wait_asynccnt 0" ::: "memory");
#endif
#endif
}

// Copy rows x (cpr*8) u16 tile, global row stride = srcStride elems, into a
// dense LDS tile.  cpr = 16-byte chunks per row.  Call sites pass literals so
// the divisions fold away.
static __device__ __forceinline__ void copy_tile(unsigned short* dst,
                                                 const unsigned short* src,
                                                 long long srcStride, int rows,
                                                 int cpr, int tid) {
  const int total = rows * cpr;
  for (int f = tid; f < total; f += 256) {
    const int r = f / cpr;
    const int c = f - r * cpr;
    copy16(src + (long long)r * srcStride + c * 8, dst + r * (cpr * 8) + c * 8);
  }
}

static __device__ __forceinline__ unsigned short f2bf(float f) {
  unsigned int u = __float_as_uint(f);
  u += 0x7FFFu + ((u >> 16) & 1u);  // round-to-nearest-even
  return (unsigned short)(u >> 16);
}
static __device__ __forceinline__ float bf2f(unsigned short h) {
  return __uint_as_float(((unsigned int)h) << 16);
}

// A-fragment (16x32 bf16, ISA 7.12.2): v0..3 K=8*half+0..7, v4..7 K=16+8*half+0..7
static __device__ __forceinline__ v16bf frag_a(const unsigned short* p, int half) {
  const uint4 lo = *(const uint4*)(p + 8 * half);
  const uint4 hi = *(const uint4*)(p + 16 + 8 * half);
  v8u u = {lo.x, lo.y, lo.z, lo.w, hi.x, hi.y, hi.z, hi.w};
  return __builtin_bit_cast(v16bf, u);
}
// B-fragment (32x16 bf16): v0..3 K=8*half+0..7, v4..7 K=16+8*half+0..7 (p = B^T row)
static __device__ __forceinline__ v16bf frag_b(const unsigned short* p, int half) {
  const uint4 lo = *(const uint4*)(p + 8 * half);
  const uint4 hi = *(const uint4*)(p + 16 + 8 * half);
  v8u u = {lo.x, lo.y, lo.z, lo.w, hi.x, hi.y, hi.z, hi.w};
  return __builtin_bit_cast(v16bf, u);
}

// split-bf16 f32 emulation: (ah+al)(bh+bl) ~= ah*bh + al*bh + ah*bl
static __device__ __forceinline__ v8f wmma_split(v16bf ah, v16bf al, v16bf bh,
                                                 v16bf bl, v8f c) {
  c = __builtin_amdgcn_wmma_f32_16x16x32_bf16(false, ah, false, bh, (short)0, c, false, false);
  c = __builtin_amdgcn_wmma_f32_16x16x32_bf16(false, al, false, bh, (short)0, c, false, false);
  c = __builtin_amdgcn_wmma_f32_16x16x32_bf16(false, ah, false, bl, (short)0, c, false, false);
  return c;
}

// ---------------------------------------------------------------------------
// One-time f32 -> split-bf16 planes, optional transpose (for weights -> [N][K])
// ---------------------------------------------------------------------------
__global__ __launch_bounds__(256) void split_kernel(
    const float* __restrict__ src, unsigned short* __restrict__ dh,
    unsigned short* __restrict__ dl, int rows, int cols, int doT) {
  const long long i = (long long)blockIdx.x * 256 + threadIdx.x;
  if (i >= (long long)rows * cols) return;
  const int r = (int)(i / cols);
  const int c = (int)(i - (long long)r * cols);
  const float v = src[i];
  const unsigned short h = f2bf(v);
  const long long o = doT ? ((long long)c * rows + r) : i;
  dh[o] = h;
  dl[o] = f2bf(v - bf2f(h));
}

// ---------------------------------------------------------------------------
// GEMM on pre-split bf16 planes: A [M][K], B^T [N][K].  Out tile 128x128,
// K step 32, double-buffered async LDS staging.  Optional fused RMSNorm over
// the 128-wide head tile; output either f32 or split-bf16 planes.
// Addressing: idx = b*batchStride + blockIdx.x*headStride + s*rowStride + col*colStride
// ---------------------------------------------------------------------------
__global__ __launch_bounds__(256) void gemm_bf16_kernel(
    const unsigned short* __restrict__ Ahg, const unsigned short* __restrict__ Alg,
    const unsigned short* __restrict__ Bhg, const unsigned short* __restrict__ Blg,
    int M, int N, int K, const float* __restrict__ gamma, int doNorm,
    int outSplit, float* __restrict__ outF, unsigned short* __restrict__ outH,
    unsigned short* __restrict__ outL, int S, long long rowStride,
    long long headStride, long long batchStride, long long colStride) {
  (void)M; (void)N;
  extern __shared__ unsigned short smem[];
  // two stages x 4 planes x [128][32] u16
  float* rowSq = (float*)(smem + 2 * 4 * 4096);

  const int tid = threadIdx.x;
  const int lane = tid & 31, wave = tid >> 5;
  const int half = lane >> 4, l16 = lane & 15;
  const int waveM = wave >> 2;  // 0..1 (64 rows)
  const int waveN = wave & 3;   // 0..3 (32 cols)
  const long long mBase = (long long)blockIdx.y * 128;
  const long long nBase = (long long)blockIdx.x * 128;

  v8f acc[4][2];
#pragma unroll
  for (int a = 0; a < 4; ++a)
#pragma unroll
    for (int b = 0; b < 2; ++b)
#pragma unroll
      for (int e = 0; e < 8; ++e) acc[a][b][e] = 0.0f;

  if (tid < 128) rowSq[tid] = 0.0f;

  auto stage = [&](int sbuf, int kt) {
    unsigned short* base = smem + sbuf * (4 * 4096);
    copy_tile(base + 0 * 4096, Ahg + mBase * K + kt, K, 128, 4, tid);
    copy_tile(base + 1 * 4096, Alg + mBase * K + kt, K, 128, 4, tid);
    copy_tile(base + 2 * 4096, Bhg + nBase * K + kt, K, 128, 4, tid);
    copy_tile(base + 3 * 4096, Blg + nBase * K + kt, K, 128, 4, tid);
  };

  stage(0, 0);
  const int nIter = K / 32;
  for (int i = 0; i < nIter; ++i) {
    async_join();
    __syncthreads();  // all waves: current stage landed, prev compute done
    if (i + 1 < nIter) stage((i + 1) & 1, (i + 1) * 32);  // overlaps compute

    const unsigned short* base = smem + (i & 1) * (4 * 4096);
    const unsigned short* Ah = base;
    const unsigned short* Al = base + 4096;
    const unsigned short* Bh = base + 2 * 4096;
    const unsigned short* Bl = base + 3 * 4096;

    v16bf bhf[2], blf[2];
#pragma unroll
    for (int nt = 0; nt < 2; ++nt) {
      const int nrow = waveN * 32 + nt * 16 + l16;
      bhf[nt] = frag_b(Bh + nrow * 32, half);
      blf[nt] = frag_b(Bl + nrow * 32, half);
    }
#pragma unroll
    for (int mt = 0; mt < 4; ++mt) {
      const int arow = waveM * 64 + mt * 16 + l16;
      const v16bf ah = frag_a(Ah + arow * 32, half);
      const v16bf al = frag_a(Al + arow * 32, half);
#pragma unroll
      for (int nt = 0; nt < 2; ++nt)
        acc[mt][nt] = wmma_split(ah, al, bhf[nt], blf[nt], acc[mt][nt]);
    }
  }

  if (doNorm) {
    __syncthreads();
#pragma unroll
    for (int mt = 0; mt < 4; ++mt)
#pragma unroll
      for (int nt = 0; nt < 2; ++nt)
#pragma unroll
        for (int r = 0; r < 8; ++r) {
          const int row = waveM * 64 + mt * 16 + r + 8 * half;
          const float v = acc[mt][nt][r];
          atomicAdd(&rowSq[row], v * v);  // ds_add_f32
        }
    __syncthreads();
  }

#pragma unroll
  for (int mt = 0; mt < 4; ++mt)
#pragma unroll
    for (int nt = 0; nt < 2; ++nt)
#pragma unroll
      for (int r = 0; r < 8; ++r) {
        const int row = waveM * 64 + mt * 16 + r + 8 * half;
        const int col = waveN * 32 + nt * 16 + l16;
        float v = acc[mt][nt][r];
        if (doNorm)
          v = gamma[col] * v * rsqrtf(rowSq[row] * (1.0f / 128.0f) + 1e-6f);
        const long long m = mBase + row;
        const int bb = (int)(m / S);
        const int ss = (int)(m % S);
        const long long idx = (long long)bb * batchStride +
                              (long long)blockIdx.x * headStride +
                              (long long)ss * rowStride +
                              (long long)col * colStride;
        if (outSplit) {
          const unsigned short hh = f2bf(v);
          outH[idx] = hh;
          outL[idx] = f2bf(v - bf2f(hh));
        } else {
          outF[idx] = v;
        }
      }
}

// ---------------------------------------------------------------------------
// Flash attention on pre-split planes.  Q [B,H,S,D], K [B,KV,S,D],
// V^T [B,KV,D,S] (so the P@V B-operand stages with straight copies).
// One WG = (b, h, 64-row q tile); key tiles of 64, double-buffered async.
// Output: split-bf16 planes [B,S,H*D] for the final projection GEMM.
// ---------------------------------------------------------------------------
__global__ __launch_bounds__(256) void attn_kernel(
    const unsigned short* __restrict__ Qhg, const unsigned short* __restrict__ Qlg,
    const unsigned short* __restrict__ Khg, const unsigned short* __restrict__ Klg,
    const unsigned short* __restrict__ Vhg, const unsigned short* __restrict__ Vlg,
    unsigned short* __restrict__ Oh, unsigned short* __restrict__ Ol) {
  extern __shared__ unsigned short smem[];
  unsigned short* Qh = smem;                  // [64][128]
  unsigned short* Ql = Qh + 8192;
  unsigned short* kvs = Ql + 8192;            // 2 stages x (Kh,Kl [64][128] ; Vth,Vtl [128][64])
  unsigned short* Ph = kvs + 2 * 32768;       // [64][64]
  unsigned short* Pl = Ph + 4096;
  float* Sbuf = (float*)(Pl + 4096);          // [64][64]
  float* mrow = Sbuf + 4096;
  float* lrow = mrow + 64;
  float* srow = lrow + 64;
  float* red  = srow + 64;                    // [64][4]

  const int tid = threadIdx.x;
  const int lane = tid & 31, wave = tid >> 5;
  const int half = lane >> 4, l16 = lane & 15;
  const int wm = wave >> 1;  // 0..3, 16 rows
  const int wn = wave & 1;   // 0..1

  const int qTile = blockIdx.x;
  const int h = blockIdx.y;
  const int b = blockIdx.z;
  const int kv = h >> 2;

  const unsigned short* Qph = Qhg + ((((long long)b * NHEADS + h) * SEQ) + (long long)qTile * 64) * HDIM;
  const unsigned short* Qpl = Qlg + ((((long long)b * NHEADS + h) * SEQ) + (long long)qTile * 64) * HDIM;
  const unsigned short* Kph = Khg + (((long long)b * NKV + kv) * SEQ) * HDIM;
  const unsigned short* Kpl = Klg + (((long long)b * NKV + kv) * SEQ) * HDIM;
  const unsigned short* Vph = Vhg + (((long long)b * NKV + kv) * HDIM) * SEQ;  // [D][S]
  const unsigned short* Vpl = Vlg + (((long long)b * NKV + kv) * HDIM) * SEQ;

  auto stageKV = [&](int sbuf, int kt) {
    unsigned short* base = kvs + sbuf * 32768;
    const long long ko = (long long)kt * 64;
    copy_tile(base + 0,     Kph + ko * HDIM, HDIM, 64, 16, tid);  // keys x d
    copy_tile(base + 8192,  Kpl + ko * HDIM, HDIM, 64, 16, tid);
    copy_tile(base + 16384, Vph + ko, SEQ, 128, 8, tid);          // d x keys
    copy_tile(base + 24576, Vpl + ko, SEQ, 128, 8, tid);
  };

  copy_tile(Qh, Qph, HDIM, 64, 16, tid);
  copy_tile(Ql, Qpl, HDIM, 64, 16, tid);
  stageKV(0, 0);
  if (tid < 64) { mrow[tid] = -3.0e38f; lrow[tid] = 0.0f; }

  v8f o[4];
#pragma unroll
  for (int nt = 0; nt < 4; ++nt)
#pragma unroll
    for (int e = 0; e < 8; ++e) o[nt][e] = 0.0f;

  const float scale = 0.08838834764831845f;  // 1/sqrt(128)

  for (int kt = 0; kt <= qTile; ++kt) {
    async_join();
    __syncthreads();
    if (kt + 1 <= qTile) stageKV((kt + 1) & 1, kt + 1);  // overlaps compute

    const unsigned short* base = kvs + (kt & 1) * 32768;
    const unsigned short* Kh = base;
    const unsigned short* Kl = base + 8192;
    const unsigned short* Vth = base + 16384;
    const unsigned short* Vtl = base + 24576;

    // S = Q K^T (64x64, K-dim 128 in 4 chunks)
    v8f sacc[2];
#pragma unroll
    for (int nt = 0; nt < 2; ++nt)
#pragma unroll
      for (int e = 0; e < 8; ++e) sacc[nt][e] = 0.0f;
#pragma unroll
    for (int kc = 0; kc < 4; ++kc) {
      const int arow = wm * 16 + l16;
      const v16bf ah = frag_a(Qh + arow * 128 + kc * 32, half);
      const v16bf al = frag_a(Ql + arow * 128 + kc * 32, half);
#pragma unroll
      for (int nt = 0; nt < 2; ++nt) {
        const int krow = wn * 32 + nt * 16 + l16;
        const v16bf bh = frag_b(Kh + krow * 128 + kc * 32, half);
        const v16bf bl = frag_b(Kl + krow * 128 + kc * 32, half);
        sacc[nt] = wmma_split(ah, al, bh, bl, sacc[nt]);
      }
    }
    const bool diag = (kt == qTile);
#pragma unroll
    for (int nt = 0; nt < 2; ++nt)
#pragma unroll
      for (int r = 0; r < 8; ++r) {
        const int row = wm * 16 + r + 8 * half;
        const int col = wn * 32 + nt * 16 + l16;
        float v = sacc[nt][r] * scale;
        if (diag && col > row) v = -3.0e38f;  // causal mask (diagonal tile)
        Sbuf[row * 64 + col] = v;
      }
    __syncthreads();

    // online softmax: partial row max (4 threads per row)
    {
      const int r = tid >> 2, seg = tid & 3;
      const float* sp = Sbuf + r * 64 + seg * 16;
      float pm = -3.0e38f;
#pragma unroll
      for (int j = 0; j < 16; ++j) pm = fmaxf(pm, sp[j]);
      red[r * 4 + seg] = pm;
    }
    __syncthreads();
    if (tid < 64) {
      const float rm = fmaxf(fmaxf(red[tid * 4 + 0], red[tid * 4 + 1]),
                             fmaxf(red[tid * 4 + 2], red[tid * 4 + 3]));
      const float mo = mrow[tid];
      const float mn = fmaxf(mo, rm);
      const float sc = __expf(mo - mn);
      mrow[tid] = mn;
      srow[tid] = sc;
      lrow[tid] *= sc;
    }
    __syncthreads();
    // P = exp(S - m_new) as split bf16 + partial row sums
    {
      const int r = tid >> 2, seg = tid & 3;
      const float mn = mrow[r];
      float ps = 0.0f;
#pragma unroll
      for (int j = 0; j < 16; ++j) {
        const int c = seg * 16 + j;
        const float p = __expf(Sbuf[r * 64 + c] - mn);
        ps += p;
        const unsigned short hh = f2bf(p);
        Ph[r * 64 + c] = hh;
        Pl[r * 64 + c] = f2bf(p - bf2f(hh));
      }
      red[r * 4 + seg] = ps;
    }
    __syncthreads();
    if (tid < 64)
      lrow[tid] += red[tid * 4 + 0] + red[tid * 4 + 1] + red[tid * 4 + 2] + red[tid * 4 + 3];
    // rescale O accumulators
#pragma unroll
    for (int nt = 0; nt < 4; ++nt)
#pragma unroll
      for (int r = 0; r < 8; ++r) {
        const int row = wm * 16 + r + 8 * half;
        o[nt][r] *= srow[row];
      }
    // O += P @ V (K-dim 64 keys in 2 chunks, N = 128 d)
#pragma unroll
    for (int kc = 0; kc < 2; ++kc) {
      const int arow = wm * 16 + l16;
      const v16bf ah = frag_a(Ph + arow * 64 + kc * 32, half);
      const v16bf al = frag_a(Pl + arow * 64 + kc * 32, half);
#pragma unroll
      for (int nt = 0; nt < 4; ++nt) {
        const int nrow = wn * 64 + nt * 16 + l16;
        const v16bf bh = frag_b(Vth + nrow * 64 + kc * 32, half);
        const v16bf bl = frag_b(Vtl + nrow * 64 + kc * 32, half);
        o[nt] = wmma_split(ah, al, bh, bl, o[nt]);
      }
    }
  }
  __syncthreads();
#pragma unroll
  for (int nt = 0; nt < 4; ++nt)
#pragma unroll
    for (int r = 0; r < 8; ++r) {
      const int row = wm * 16 + r + 8 * half;
      const int col = wn * 64 + nt * 16 + l16;
      const float v = o[nt][r] / lrow[row];
      const long long idx =
          (((long long)b * SEQ) + (long long)qTile * 64 + row) * (NHEADS * HDIM) +
          (long long)h * HDIM + col;
      const unsigned short hh = f2bf(v);
      Oh[idx] = hh;
      Ol[idx] = f2bf(v - bf2f(hh));
    }
}

// ---------------------------------------------------------------------------
extern "C" void kernel_launch(void* const* d_in, const int* in_sizes, int n_in,
                              void* d_out, int out_size, void* d_ws, size_t ws_size,
                              hipStream_t stream) {
  (void)in_sizes; (void)n_in; (void)out_size; (void)ws_size;
  const float* x  = (const float*)d_in[0];
  const float* Wq = (const float*)d_in[1];
  const float* Wk = (const float*)d_in[2];
  const float* Wv = (const float*)d_in[3];
  const float* Wo = (const float*)d_in[4];
  const float* gq = (const float*)d_in[5];
  const float* gk = (const float*)d_in[6];
  float* out = (float*)d_out;

  // workspace: split-bf16 planes (~152 MB)
  unsigned short* p = (unsigned short*)d_ws;
  const size_t nX  = (size_t)NB * SEQ * HIDDEN;          // x / attn-out planes
  const size_t nWq = (size_t)HIDDEN * NHEADS * HDIM;
  const size_t nWk = (size_t)HIDDEN * NKV * HDIM;
  const size_t nQ  = (size_t)NB * NHEADS * SEQ * HDIM;
  const size_t nK  = (size_t)NB * NKV * SEQ * HDIM;
  unsigned short* xh  = p; p += nX;   unsigned short* xl  = p; p += nX;
  unsigned short* wqh = p; p += nWq;  unsigned short* wql = p; p += nWq;
  unsigned short* wkh = p; p += nWk;  unsigned short* wkl = p; p += nWk;
  unsigned short* wvh = p; p += nWk;  unsigned short* wvl = p; p += nWk;
  unsigned short* woh = p; p += nWq;  unsigned short* wol = p; p += nWq;
  unsigned short* qh  = p; p += nQ;   unsigned short* ql  = p; p += nQ;
  unsigned short* kh  = p; p += nK;   unsigned short* kl  = p; p += nK;
  unsigned short* vth = p; p += nK;   unsigned short* vtl = p; p += nK;
  unsigned short* oh  = p; p += nX;   unsigned short* ol  = p; p += nX;

  dim3 blk(256, 1, 1);
  const int M = NB * SEQ;
  const size_t gemmLds = 2 * 4 * 4096 * sizeof(unsigned short) + 128 * sizeof(float);
  const size_t attnLds = (size_t)(2 * 8192 + 2 * 32768 + 2 * 4096) * sizeof(unsigned short)
                       + (size_t)(4096 + 3 * 64 + 256) * sizeof(float);

  // one-time operand splitting (weights transposed to [N][K])
  split_kernel<<<dim3((unsigned)((nX + 255) / 256)), blk, 0, stream>>>(x, xh, xl, M, HIDDEN, 0);
  split_kernel<<<dim3((unsigned)((nWq + 255) / 256)), blk, 0, stream>>>(Wq, wqh, wql, HIDDEN, NHEADS * HDIM, 1);
  split_kernel<<<dim3((unsigned)((nWk + 255) / 256)), blk, 0, stream>>>(Wk, wkh, wkl, HIDDEN, NKV * HDIM, 1);
  split_kernel<<<dim3((unsigned)((nWk + 255) / 256)), blk, 0, stream>>>(Wv, wvh, wvl, HIDDEN, NKV * HDIM, 1);
  split_kernel<<<dim3((unsigned)((nWq + 255) / 256)), blk, 0, stream>>>(Wo, woh, wol, NHEADS * HDIM, HIDDEN, 1);

  // Q = rmsnorm(x @ Wq)*gq -> [B,H,S,D] split planes
  gemm_bf16_kernel<<<dim3(NHEADS, M / 128), blk, gemmLds, stream>>>(
      xh, xl, wqh, wql, M, NHEADS * HDIM, HIDDEN, gq, 1, 1, nullptr, qh, ql, SEQ,
      (long long)HDIM, (long long)SEQ * HDIM, (long long)NHEADS * SEQ * HDIM, 1LL);
  // K = rmsnorm(x @ Wk)*gk -> [B,KV,S,D]
  gemm_bf16_kernel<<<dim3(NKV, M / 128), blk, gemmLds, stream>>>(
      xh, xl, wkh, wkl, M, NKV * HDIM, HIDDEN, gk, 1, 1, nullptr, kh, kl, SEQ,
      (long long)HDIM, (long long)SEQ * HDIM, (long long)NKV * SEQ * HDIM, 1LL);
  // V = x @ Wv -> transposed planes [B,KV,D,S]  (colStride = S, rowStride = 1)
  gemm_bf16_kernel<<<dim3(NKV, M / 128), blk, gemmLds, stream>>>(
      xh, xl, wvh, wvl, M, NKV * HDIM, HIDDEN, nullptr, 0, 1, nullptr, vth, vtl, SEQ,
      1LL, (long long)HDIM * SEQ, (long long)NKV * HDIM * SEQ, (long long)SEQ);
  // causal flash attention -> [B,S,H*D] split planes
  attn_kernel<<<dim3(SEQ / 64, NHEADS, NB), blk, attnLds, stream>>>(
      qh, ql, kh, kl, vth, vtl, oh, ol);
  // out = attn @ Wo  (f32 output)
  gemm_bf16_kernel<<<dim3(NHEADS, M / 128), blk, gemmLds, stream>>>(
      oh, ol, woh, wol, M, HIDDEN, NHEADS * HDIM, nullptr, 0, 0, out, nullptr, nullptr, SEQ,
      (long long)HIDDEN, 128LL, (long long)SEQ * HIDDEN, 1LL);
}
